// MultiHeadSelfAttention_68805376081941
// MI455X (gfx1250) — compile-verified, matrix operand order checked by reading, not run on hardware
//
#include <hip/hip_runtime.h>

// ---------------------------------------------------------------------------
// CDNA5 (gfx1250) wave32 WMMA multi-head attention.
//   B=4, S=2048, D=1024, H=16, d=64.  bf16 WMMA, f32 accumulate.
//   LDS block-tiling with async global->LDS staging, double buffered.
// ---------------------------------------------------------------------------

typedef __attribute__((ext_vector_type(16))) __bf16 bf16x16;
typedef __attribute__((ext_vector_type(8)))  float  f32x8;
typedef __attribute__((ext_vector_type(4)))  int    i32x4;

static __device__ __forceinline__ unsigned short bf16b(float f) {
  unsigned int u = __builtin_bit_cast(unsigned int, f);
  u += 0x7fffu + ((u >> 16) & 1u);          // round-to-nearest-even
  return (unsigned short)(u >> 16);
}
static __device__ __forceinline__ unsigned int pack2(float a, float b) {
  return (unsigned int)bf16b(a) | ((unsigned int)bf16b(b) << 16);
}

union LdFrag { bf16x16 v; uint4 q[2]; };

// Fragment gather (16x32 bf16 A/B layout), from a row pointer:
//   lane L: row baked into rowp by caller, koff = 8*(L>>4)
//   elements 0..7  = row[koff .. koff+7]
//   elements 8..15 = row[16+koff .. 16+koff+7]
static __device__ __forceinline__ bf16x16 lds_frag(const __bf16* rowp, int koff) {
  LdFrag f;
  f.q[0] = *(const uint4*)(rowp + koff);
  f.q[1] = *(const uint4*)(rowp + 16 + koff);
  return f.v;
}
static __device__ __forceinline__ bf16x16 gbl_frag(const __bf16* rowp, int k0, int koff) {
  LdFrag f;
  f.q[0] = *(const uint4*)(rowp + k0 + koff);
  f.q[1] = *(const uint4*)(rowp + k0 + 16 + koff);
  return f.v;
}

static __device__ __forceinline__ f32x8 wmma_bf16(bf16x16 a, bf16x16 b, f32x8 c) {
  return __builtin_amdgcn_wmma_f32_16x16x32_bf16(false, a, false, b, (short)0, c,
                                                 false, false);
}

// ---- async global -> LDS (16B per lane), with portable fallback ------------
#if defined(__has_builtin)
#if __has_builtin(__builtin_amdgcn_global_load_async_to_lds_b128)
#define ASYNC_LDS_COPY 1
#endif
#endif

static __device__ __forceinline__ void cp16(void* lds, const void* g) {
#ifdef ASYNC_LDS_COPY
  __builtin_amdgcn_global_load_async_to_lds_b128(
      (__attribute__((address_space(1))) i32x4*)(g),
      (__attribute__((address_space(3))) i32x4*)(lds), 0, 0);
#else
  *(uint4*)lds = *(const uint4*)g;
#endif
}
static __device__ __forceinline__ void async_wait() {
#ifdef ASYNC_LDS_COPY
  asm volatile("s_wait_asynccnt 0" ::: "memory");
#endif
}

// ---------------------------------------------------------------------------
// f32 -> bf16 bulk convert (4 elems/thread)
// ---------------------------------------------------------------------------
__global__ void cvt4_kernel(const float* __restrict__ src, __bf16* __restrict__ dst,
                            int n4) {
  int i = blockIdx.x * blockDim.x + threadIdx.x;
  if (i >= n4) return;
  float4 f = ((const float4*)src)[i];
  uint2 p;
  p.x = pack2(f.x, f.y);
  p.y = pack2(f.z, f.w);
  ((uint2*)dst)[i] = p;
}

// ---------------------------------------------------------------------------
// Transpose + convert a 1024x1024 f32 weight: Wt[n][k] = W[k][n]  (bf16)
// ---------------------------------------------------------------------------
__global__ void cvtT_kernel(const float* __restrict__ W, __bf16* __restrict__ Wt) {
  __shared__ float tile[32][33];
  const int tx = threadIdx.x, ty = threadIdx.y;
  const int n0 = blockIdx.x << 5, k0 = blockIdx.y << 5;
#pragma unroll
  for (int i = 0; i < 4; ++i)
    tile[ty + 8 * i][tx] = W[(size_t)(k0 + ty + 8 * i) * 1024 + n0 + tx];
  __syncthreads();
#pragma unroll
  for (int i = 0; i < 4; ++i) {
    unsigned short v = bf16b(tile[tx][ty + 8 * i]);
    ((unsigned short*)Wt)[(size_t)(n0 + ty + 8 * i) * 1024 + k0 + tx] = v;
  }
}

// ---------------------------------------------------------------------------
// LDS-tiled bf16 WMMA GEMM:  D[i][j] = sum_k A[i][k] * B[j][k]   (K = 1024)
// Block tile: 64(i) x 128(j), k-step 32, double-buffered async LDS staging.
// 8 waves; each wave computes 32x32 (2x2 WMMA tiles).
// mode 0/1: Q/K  -> bf16 [B][H][S][64]   (i = embed-n, j = token-m)
// mode 2  : V    -> bf16 Vt [B][H][64][S] (i = token-m, j = embed-n)
// mode 3  : proj -> f32 [8192][1024]      (i = embed-n, j = token-m)
// ---------------------------------------------------------------------------
__global__ __launch_bounds__(256)
void gemm_bf16_kernel(const __bf16* __restrict__ A, const __bf16* __restrict__ B,
                      const float* __restrict__ bias, void* __restrict__ outp,
                      int IM, int JN, int mode) {
  __shared__ __attribute__((aligned(16))) __bf16 As[2][64][40];   // padded rows
  __shared__ __attribute__((aligned(16))) __bf16 Bs[2][128][40];

  const int tid = threadIdx.x;
  const int lane = tid & 31;
  const int w = tid >> 5;
  const int lr = lane & 15, koff = (lane >> 4) << 3;
  const int wi = w >> 2, wj = w & 3;

  const int jt = JN >> 7;
  const int bi = (int)blockIdx.x / jt;
  const int j0 = ((int)blockIdx.x - bi * jt) << 7;
  const int i0 = bi << 6;

  // staging: A 64x32 (1 chunk/thread), B 128x32 (2 chunks/thread)
  const int ar = tid >> 2, ac = (tid & 3) << 3;
  const __bf16* Ag  = A + (size_t)(i0 + ar) * 1024 + ac;
  const __bf16* Bg0 = B + (size_t)(j0 + ar) * 1024 + ac;
  const __bf16* Bg1 = B + (size_t)(j0 + 64 + ar) * 1024 + ac;

  auto stage = [&](int buf, int k0) {
    cp16(&As[buf][ar][ac], Ag + k0);
    cp16(&Bs[buf][ar][ac], Bg0 + k0);
    cp16(&Bs[buf][64 + ar][ac], Bg1 + k0);
  };

  const f32x8 z = {0.f, 0.f, 0.f, 0.f, 0.f, 0.f, 0.f, 0.f};
  f32x8 acc00 = z, acc01 = z, acc10 = z, acc11 = z;

  const int ra0 = (wi << 5) + lr, ra1 = ra0 + 16;
  const int rb0 = (wj << 5) + lr, rb1 = rb0 + 16;

  stage(0, 0);
  async_wait();
  __syncthreads();

  for (int ks = 0; ks < 32; ++ks) {
    const int cur = ks & 1;
    if (ks + 1 < 32) stage(cur ^ 1, (ks + 1) << 5);
    bf16x16 a0 = lds_frag(&As[cur][ra0][0], koff);
    bf16x16 a1 = lds_frag(&As[cur][ra1][0], koff);
    bf16x16 b0 = lds_frag(&Bs[cur][rb0][0], koff);
    bf16x16 b1 = lds_frag(&Bs[cur][rb1][0], koff);
    acc00 = wmma_bf16(a0, b0, acc00);
    acc01 = wmma_bf16(a0, b1, acc01);
    acc10 = wmma_bf16(a1, b0, acc10);
    acc11 = wmma_bf16(a1, b1, acc11);
    async_wait();
    __syncthreads();
  }

  f32x8 accs[2][2] = {{acc00, acc01}, {acc10, acc11}};
#pragma unroll
  for (int ti = 0; ti < 2; ++ti) {
    const int ibase = i0 + (wi << 5) + ti * 16;
#pragma unroll
    for (int tj = 0; tj < 2; ++tj) {
      const int jbase = j0 + (wj << 5) + tj * 16;
      f32x8 acc = accs[ti][tj];
      if (mode <= 1) {
        __bf16* O = (__bf16*)outp;
        const int h = ibase >> 6;
        const int e0 = (ibase & 63) + koff;
        const float* bp = bias + ibase + koff;
        int m = jbase + lr;
        int b = m >> 11, s = m & 2047;
        uint4 pkt;
        pkt.x = pack2(acc[0] + bp[0], acc[1] + bp[1]);
        pkt.y = pack2(acc[2] + bp[2], acc[3] + bp[3]);
        pkt.z = pack2(acc[4] + bp[4], acc[5] + bp[5]);
        pkt.w = pack2(acc[6] + bp[6], acc[7] + bp[7]);
        *(uint4*)(O + ((((size_t)b * 16 + h) * 2048 + s) * 64 + e0)) = pkt;
      } else if (mode == 2) {
        __bf16* O = (__bf16*)outp;
        const int b = ibase >> 11;
        const int s0 = (ibase & 2047) + koff;
        const int h = jbase >> 6;
        const int e = (jbase & 63) + lr;
        float bv = bias[jbase + lr];
        uint4 pkt;
        pkt.x = pack2(acc[0] + bv, acc[1] + bv);
        pkt.y = pack2(acc[2] + bv, acc[3] + bv);
        pkt.z = pack2(acc[4] + bv, acc[5] + bv);
        pkt.w = pack2(acc[6] + bv, acc[7] + bv);
        *(uint4*)(O + ((((size_t)b * 16 + h) * 64 + e) * 2048 + s0)) = pkt;
      } else {
        float* O = (float*)outp;
        const float* bp = bias + ibase + koff;
        int m = jbase + lr;
        float4 v0, v1;
        v0.x = acc[0] + bp[0]; v0.y = acc[1] + bp[1];
        v0.z = acc[2] + bp[2]; v0.w = acc[3] + bp[3];
        v1.x = acc[4] + bp[4]; v1.y = acc[5] + bp[5];
        v1.z = acc[6] + bp[6]; v1.w = acc[7] + bp[7];
        float* dst = O + (size_t)m * 1024 + ibase + koff;
        *(float4*)dst = v0;
        *(float4*)(dst + 4) = v1;
      }
    }
  }
}

// ---------------------------------------------------------------------------
// Flash attention. Block = 8 waves sharing one (b,h); each wave owns a
// 16-query tile. kv blocks of 32 staged (K 32x64, V 64x32) into LDS once per
// block and consumed by all waves. S^T = K.Q^T is computed so the exp'd
// probability tile is already in WMMA B-operand layout for O^T = Vt . P^T
// (pure in-lane bf16 packing, no transpose). Writes Cb[b][s][h*64+e] (bf16).
// ---------------------------------------------------------------------------
__global__ __launch_bounds__(256)
void attn_kernel(const __bf16* __restrict__ Qh, const __bf16* __restrict__ Kh,
                 const __bf16* __restrict__ Vt, __bf16* __restrict__ Cb) {
  __shared__ __attribute__((aligned(16))) __bf16 Ks[2][32][72];
  __shared__ __attribute__((aligned(16))) __bf16 Vs[2][64][40];

  const int tid = threadIdx.x;
  const int w = tid >> 5;
  const int lane = tid & 31;
  const int lr = lane & 15, koff = (lane >> 4) << 3;

  const int bh = (int)blockIdx.x >> 4;                 // 16 q-groups per (b,h)
  const int q0 = ((((int)blockIdx.x & 15) << 3) + w) << 4;

  const __bf16* Qp = Qh + ((size_t)bh * 2048 + q0) * 64 + (size_t)lr * 64;
  const __bf16* Kp = Kh + (size_t)bh * 2048 * 64;
  const __bf16* Vp = Vt + (size_t)bh * 64 * 2048;

  // staging coords: K 32x64 (1 chunk/thread), V 64x32 (1 chunk/thread)
  const int kr = tid >> 3, kc = (tid & 7) << 3;
  const int vr = tid >> 2, vc = (tid & 3) << 3;
  const __bf16* Kg = Kp + (size_t)kr * 64 + kc;
  const __bf16* Vg = Vp + (size_t)vr * 2048 + vc;

  auto stage = [&](int buf, int n0) {
    cp16(&Ks[buf][kr][kc], Kg + (size_t)n0 * 64);
    cp16(&Vs[buf][vr][vc], Vg + n0);
  };

  const bf16x16 qf0 = gbl_frag(Qp, 0, koff);    // d = 0..31 (B-operand)
  const bf16x16 qf1 = gbl_frag(Qp, 32, koff);   // d = 32..63

  const f32x8 z = {0.f, 0.f, 0.f, 0.f, 0.f, 0.f, 0.f, 0.f};
  f32x8 o0 = z, o1 = z, o2 = z, o3 = z;         // O^T tiles: e-blocks 0..3
  float m_run = -3.0e38f, l_run = 0.f;

  stage(0, 0);
  async_wait();
  __syncthreads();

  for (int ks = 0; ks < 64; ++ks) {
    const int cur = ks & 1;
    if (ks + 1 < 64) stage(cur ^ 1, (ks + 1) << 5);

    f32x8 s0 = z, s1 = z;                       // S^T tiles (rows n, cols m)
    s0 = wmma_bf16(lds_frag(&Ks[cur][lr][0], koff),       qf0, s0);
    s0 = wmma_bf16(lds_frag(&Ks[cur][lr][32], koff),      qf1, s0);
    s1 = wmma_bf16(lds_frag(&Ks[cur][16 + lr][0], koff),  qf0, s1);
    s1 = wmma_bf16(lds_frag(&Ks[cur][16 + lr][32], koff), qf1, s1);

    float t0[8], t1[8];
    float mx = -3.0e38f;
#pragma unroll
    for (int r = 0; r < 8; ++r) {
      t0[r] = s0[r] * 0.125f;                   // 1/sqrt(64)
      t1[r] = s1[r] * 0.125f;
      mx = fmaxf(mx, fmaxf(t0[r], t1[r]));
    }
    mx = fmaxf(mx, __shfl_xor(mx, 16, 32));     // other half of n-range
    float m_new = fmaxf(m_run, mx);
    float alpha = __expf(m_run - m_new);

    union { bf16x16 v; unsigned short us[16]; } pb;
    float lsum = 0.f;
#pragma unroll
    for (int r = 0; r < 8; ++r) {
      float e0 = __expf(t0[r] - m_new);
      float e1 = __expf(t1[r] - m_new);
      lsum += e0 + e1;
      pb.us[r]     = bf16b(e0);                 // K = n0+hi*8+r     (tile 0)
      pb.us[8 + r] = bf16b(e1);                 // K = n0+16+hi*8+r  (tile 1)
    }
    lsum += __shfl_xor(lsum, 16, 32);
    l_run = l_run * alpha + lsum;
    m_run = m_new;
#pragma unroll
    for (int r = 0; r < 8; ++r) {
      o0[r] *= alpha; o1[r] *= alpha; o2[r] *= alpha; o3[r] *= alpha;
    }
    o0 = wmma_bf16(lds_frag(&Vs[cur][lr][0], koff),      pb.v, o0);
    o1 = wmma_bf16(lds_frag(&Vs[cur][16 + lr][0], koff), pb.v, o1);
    o2 = wmma_bf16(lds_frag(&Vs[cur][32 + lr][0], koff), pb.v, o2);
    o3 = wmma_bf16(lds_frag(&Vs[cur][48 + lr][0], koff), pb.v, o3);

    async_wait();
    __syncthreads();
  }

  const float inv = 1.f / l_run;
  const int b = bh >> 4, h = bh & 15;
  __bf16* crow = Cb + ((size_t)b * 2048 + q0 + lr) * 1024 + h * 64 + koff;
  f32x8 oo[4] = {o0, o1, o2, o3};
#pragma unroll
  for (int t = 0; t < 4; ++t) {
    uint4 pkt;
    pkt.x = pack2(oo[t][0] * inv, oo[t][1] * inv);
    pkt.y = pack2(oo[t][2] * inv, oo[t][3] * inv);
    pkt.z = pack2(oo[t][4] * inv, oo[t][5] * inv);
    pkt.w = pack2(oo[t][6] * inv, oo[t][7] * inv);
    *(uint4*)(crow + t * 16) = pkt;
  }
}

// ---------------------------------------------------------------------------
extern "C" void kernel_launch(void* const* d_in, const int* in_sizes, int n_in,
                              void* d_out, int out_size, void* d_ws, size_t ws_size,
                              hipStream_t stream) {
  (void)in_sizes; (void)n_in; (void)out_size; (void)ws_size;
  const float* X  = (const float*)d_in[0];
  const float* Wq = (const float*)d_in[1];
  const float* bq = (const float*)d_in[2];
  const float* Wk = (const float*)d_in[3];
  const float* bk = (const float*)d_in[4];
  const float* Wv = (const float*)d_in[5];
  const float* bv = (const float*)d_in[6];
  const float* Wo = (const float*)d_in[7];
  const float* bo = (const float*)d_in[8];
  float* out = (float*)d_out;

  char* ws = (char*)d_ws;
  __bf16* Xb  = (__bf16*)(ws);                       // 16 MiB (reused as Cb)
  __bf16* Wqt = (__bf16*)(ws + (16u << 20));         // 2 MiB each
  __bf16* Wkt = (__bf16*)(ws + (18u << 20));
  __bf16* Wvt = (__bf16*)(ws + (20u << 20));
  __bf16* Wot = (__bf16*)(ws + (22u << 20));
  __bf16* Qh  = (__bf16*)(ws + (24u << 20));         // 16 MiB
  __bf16* Kh  = (__bf16*)(ws + (40u << 20));         // 16 MiB
  __bf16* Vt  = (__bf16*)(ws + (56u << 20));         // 16 MiB (ends 72 MiB)
  __bf16* Cb  = Xb;                                  // safe: stream-ordered

  cvt4_kernel<<<8192, 256, 0, stream>>>(X, Xb, 2097152);

  dim3 tb(32, 8), tg(32, 32);
  cvtT_kernel<<<tg, tb, 0, stream>>>(Wq, Wqt);
  cvtT_kernel<<<tg, tb, 0, stream>>>(Wk, Wkt);
  cvtT_kernel<<<tg, tb, 0, stream>>>(Wv, Wvt);
  cvtT_kernel<<<tg, tb, 0, stream>>>(Wo, Wot);

  gemm_bf16_kernel<<<1024, 256, 0, stream>>>(Wqt, Xb, bq, Qh, 1024, 8192, 0);
  gemm_bf16_kernel<<<1024, 256, 0, stream>>>(Wkt, Xb, bk, Kh, 1024, 8192, 1);
  gemm_bf16_kernel<<<1024, 256, 0, stream>>>(Xb, Wvt, bv, Vt, 8192, 1024, 2);

  attn_kernel<<<1024, 256, 0, stream>>>(Qh, Kh, Vt, Cb);

  gemm_bf16_kernel<<<1024, 256, 0, stream>>>(Wot, Cb, bo, out, 1024, 8192, 3);
}
